// RNN_7533372637512
// MI455X (gfx1250) — matrix operand order repaired and checked by reading, not compiled, not run on hardware
//
#include <hip/hip_runtime.h>
#include <stdint.h>

// ---------------------------------------------------------------------------
// MI455X (gfx1250) implementation of the sequential RNN wavefunction sampler.
//
// The reference is ONE sequential chain of 32*2048 = 65536 RNN cell steps
// (hidden state carried across both the sample axis and the spin axis).
// Each step = 5 matvecs of 512x512 -> latency/per-step-bandwidth bound.
//
//   * bf16 weights, converted once per launch into workspace (prep kernel).
//   * 16-workgroup cluster, one WG (64 thr = 2 waves) per WGP; each WG owns a
//     32-row slice of every matrix, held in 160 KB of LDS for the whole
//     2048-step inner loop of a spin (weight-stationary).
//   * Weight staging via the Tensor Data Mover (TENSOR_LOAD_TO_LDS, one D#
//     per 32 KB slice, TENSORcnt-tracked), VALU copy-loop fallback.
//   * matvec via v_wmma_f32_16x16x32_bf16: h broadcast into all 16 A rows,
//     B tile = 16 weight rows x 32 k from LDS; each wave owns 16 outputs and
//     accumulates K in its f32 accumulator -> no cross-lane reductions.
//   * h vectors (1 KB bf16) ping-pong through L2; cluster split-barriers
//     (s_barrier_signal/-wait -3) + fences order the phases.
//     NOTE: dispatch must launch the 16 WGs as one workgroup-cluster; with a
//     plain dispatch the cluster barrier degrades to S_NOP per the ISA.
//   * dense head (2x512) + softmax on wave 0 of WG 0; final product-over-spins
//     reduction in a tiny epilogue kernel.
//
// Workspace layout (bytes), requires ~84.2 MB:
//   [0,            16 MB)  Whh0 bf16  [32][512][512]
//   [16 MB,        48 MB)  Wih  bf16  [32][2][512][512]
//   [48 MB,        80 MB)  Whh  bf16  [32][2][512][512]
//   [80 MB, +6144B      )  hbuf bf16  [3 layers][2 pingpong][512]
//   [80 MB+8192, +256KB )  pbuf f32   [2048][32]
// ---------------------------------------------------------------------------

typedef uint16_t u16;
typedef uint32_t u32;

typedef __attribute__((ext_vector_type(16))) __bf16 v16bf;
typedef __attribute__((ext_vector_type(8)))  float  v8f;
typedef __attribute__((ext_vector_type(4)))  unsigned int v4u;
typedef __attribute__((ext_vector_type(8)))  int    v8i;
typedef __attribute__((ext_vector_type(4)))  int    v4i;

#define NSPIN 32
#define SEQ   2048
#define HID   512
#define NWG   16
#define ROWS_PER_WG   32
#define ROWS_PER_WAVE 16
#define SLICE_HALVES  (ROWS_PER_WG * HID)     // 16384 halves = 32 KB / matrix slice
#define SLICE_BYTES   (SLICE_HALVES * 2)

#if defined(__has_builtin)
#if __has_builtin(__builtin_amdgcn_tensor_load_to_lds)
#define HAVE_TDM 1
#endif
#if __has_builtin(__builtin_amdgcn_s_wait_tensorcnt)
#define HAVE_TENSORCNT_BUILTIN 1
#endif
#endif

__device__ __forceinline__ u16 f2bf(float f) {
  u32 u = __builtin_bit_cast(u32, f);
  u32 r = u + 0x7FFFu + ((u >> 16) & 1u);     // round-to-nearest-even
  return (u16)(r >> 16);
}
__device__ __forceinline__ float bf2f(u16 h) {
  u32 u = ((u32)h) << 16;
  return __builtin_bit_cast(float, u);
}

union ABv { v16bf v; uint4 q[2]; };

// One 16-output tile of a 512-K matvec, accumulated into c.
// hvec: global bf16 vector (len 512).  Wlds: LDS slice (32 rows x 512 k).
// A (k order chosen here, not the ISA table's): lanes 0-15 hold K=[0..15]; lanes 16-31 hold
// K=[16..31] of their row; h is broadcast into every A row (harmless:
// all D rows equal, we read row 0 == c[0]).
// B (same k order as A): lane n holds column n, K 0..15; lane n+16 K 16..31,
// i.e. 16 contiguous k of weight row (tileRow0+n)  -> two ds_load_b128.
__device__ __forceinline__ v8f matvec_tile_acc(const u16* __restrict__ hvec,
                                               const u16* Wlds,
                                               int waveLocalRow0,
                                               int lane, v8f c) {
  const int g = lane >> 4;        // K-half select
  const int n = lane & 15;        // output column within tile
#pragma unroll 4
  for (int k0 = 0; k0 < HID; k0 += 32) {
    ABv a, b;
    a.q[0] = *(const uint4*)(hvec + k0 +     g * 16);
    a.q[1] = *(const uint4*)(hvec + k0 + 8 + g * 16);
    const u16* bp = Wlds + ((waveLocalRow0 + n) * HID + k0 + g * 16);
    b.q[0] = *(const uint4*)(bp);
    b.q[1] = *(const uint4*)(bp + 8);
    c = __builtin_amdgcn_wmma_f32_16x16x32_bf16(false, a.v, false, b.v,
                                                (short)0, c, false, false);
  }
  return c;
}

__device__ __forceinline__ void cluster_sync() {
  __syncthreads();
  __threadfence();                         // release: WGP$ -> L2
  __builtin_amdgcn_s_cluster_barrier();    // s_barrier_signal -3 ; s_barrier_wait -3
  __threadfence();                         // acquire: invalidate stale lines
}

#if defined(HAVE_TDM)
// Tensor Data Mover: one descriptor moves a flat 32 KB weight slice
// (16384 x 2-byte elements as a 16384x1 2D tile) from global to LDS.
// D# packing per CDNA5 ISA.
__device__ __forceinline__ void tdm_stage(unsigned lds_byte_off,
                                          const u16* __restrict__ src) {
  unsigned long long ga = (unsigned long long)(uintptr_t)src;
  v4u g0 = { 1u,                                          // count=1, user mode
             lds_byte_off,                                // [63:32] lds_addr
             (unsigned)(ga & 0xFFFFFFFFull),              // global_addr[31:0]
             (unsigned)((ga >> 32) & 0x01FFFFFFull)       // global_addr[56:32]
               | 0x80000000u };                           // type=2 ("image")
  v8i g1 = { (int)0x00010000,   // wg_mask=0, data_size=1 (2 bytes)
             (int)0x40000000,   // tensor_dim0[15:0]=16384 in bits[31:16]
             (int)0x00010000,   // tensor_dim0[31:16]=0 | tensor_dim1=1
             (int)0x40000000,   // tensor_dim1 hi=0 | tile_dim0=16384
             (int)0x00000001,   // tile_dim1=1, tile_dim2=0
             (int)0x00004000,   // tensor_dim0_stride lo32 = 16384
             (int)0x40000000,   // stride0 hi=0 | tensor_dim1_stride[15:0]=16384
             0 };               // tensor_dim1_stride[47:16]=0
  v4i z4 = {0, 0, 0, 0};
#if (__clang_major__ >= 23)
  v8i z8 = {0, 0, 0, 0, 0, 0, 0, 0};
  __builtin_amdgcn_tensor_load_to_lds(g0, g1, z4, z4, z8, 0);
#else
  __builtin_amdgcn_tensor_load_to_lds(g0, g1, z4, z4, 0);
#endif
}
__device__ __forceinline__ void tdm_wait() {
#if defined(HAVE_TENSORCNT_BUILTIN)
  __builtin_amdgcn_s_wait_tensorcnt(0);
#else
  asm volatile("s_wait_tensorcnt 0x0" ::: "memory");
#endif
}
#endif  // HAVE_TDM

__device__ __forceinline__ void stage_slice(u16* dst_lds, const u16* __restrict__ src,
                                            int tid) {
  const uint4* s = (const uint4*)src;      // slice is 32 contiguous rows -> flat copy
  uint4* d = (uint4*)dst_lds;
  for (int i = tid; i < SLICE_HALVES / 8; i += 64) d[i] = s[i];
}

// ---------------------------------------------------------------------------
__global__ void convert_weights_bf16(const float* __restrict__ whh0,
                                     const float* __restrict__ wih,
                                     const float* __restrict__ whh,
                                     u16* __restrict__ o0, u16* __restrict__ o1,
                                     u16* __restrict__ o2) {
  long long idx = (long long)blockIdx.x * blockDim.x + threadIdx.x;
  const long long A = (long long)NSPIN * HID * HID;           // 8,388,608
  const long long B = (long long)NSPIN * 2 * HID * HID;       // 16,777,216
  if (idx < A)              o0[idx]         = f2bf(whh0[idx]);
  else if (idx < A + B)     o1[idx - A]     = f2bf(wih[idx - A]);
  else if (idx < A + 2 * B) o2[idx - A - B] = f2bf(whh[idx - A - B]);
}

__global__ void zero_hbuf(u16* hbuf) {
  int i = blockIdx.x * blockDim.x + threadIdx.x;
  if (i < 3 * 2 * HID) hbuf[i] = 0;
}

// ---------------------------------------------------------------------------
__global__ __launch_bounds__(64) void rnn_persistent(
    const int*   __restrict__ samples,  // [2048][32]
    const float* __restrict__ Wih0,     // [32][512][2]
    const float* __restrict__ bih0,     // [32][512]
    const float* __restrict__ bhh0,     // [32][512]
    const float* __restrict__ bihL,     // [32][2][512]
    const float* __restrict__ bhhL,     // [32][2][512]
    const float* __restrict__ Wd,       // [32][2][512]
    const float* __restrict__ bd,       // [32][2]
    const u16*   __restrict__ whh0b,    // bf16 [32][512][512]
    const u16*   __restrict__ wihb,     // bf16 [32][2][512][512]
    const u16*   __restrict__ whhb,     // bf16 [32][2][512][512]
    u16*         __restrict__ hbuf,     // bf16 [3][2][512]
    float*       __restrict__ pbuf) {   // f32  [2048][32]
  extern __shared__ u16 lds[];          // 5 * 32 KB = 160 KB dynamic LDS
  u16* ldsWhh0 = lds;
  u16* ldsWih1 = lds + 1 * SLICE_HALVES;
  u16* ldsWhh1 = lds + 2 * SLICE_HALVES;
  u16* ldsWih2 = lds + 3 * SLICE_HALVES;
  u16* ldsWhh2 = lds + 4 * SLICE_HALVES;

  const int wg   = blockIdx.x;          // 0..15 (one WGP each; cluster member)
  const int tid  = threadIdx.x;         // 0..63
  const int lane = tid & 31;
  const int wave = tid >> 5;            // 0 or 1
  const int row  = wg * ROWS_PER_WG + wave * ROWS_PER_WAVE + (lane & 15);

  for (int spin = 0; spin < NSPIN; ++spin) {
    const u16* srcWhh0 = whh0b + ((size_t)spin * HID + wg * ROWS_PER_WG) * HID;
    const u16* srcWih1 = wihb  + (((size_t)spin * 2 + 0) * HID + wg * ROWS_PER_WG) * HID;
    const u16* srcWhh1 = whhb  + (((size_t)spin * 2 + 0) * HID + wg * ROWS_PER_WG) * HID;
    const u16* srcWih2 = wihb  + (((size_t)spin * 2 + 1) * HID + wg * ROWS_PER_WG) * HID;
    const u16* srcWhh2 = whhb  + (((size_t)spin * 2 + 1) * HID + wg * ROWS_PER_WG) * HID;

    // ---- stage this spin's 160 KB weight slice into LDS --------------------
    __syncthreads();   // prior spin's waves done reading LDS
#if defined(HAVE_TDM)
    if (tid == 0) {    // wave 0 issues 5 TDM descriptors (EXEC-independent)
      tdm_stage(0 * SLICE_BYTES, srcWhh0);
      tdm_stage(1 * SLICE_BYTES, srcWih1);
      tdm_stage(2 * SLICE_BYTES, srcWhh1);
      tdm_stage(3 * SLICE_BYTES, srcWih2);
      tdm_stage(4 * SLICE_BYTES, srcWhh2);
      tdm_wait();      // s_wait_tensorcnt 0
    }
#else
    stage_slice(ldsWhh0, srcWhh0, tid);
    stage_slice(ldsWih1, srcWih1, tid);
    stage_slice(ldsWhh1, srcWhh1, tid);
    stage_slice(ldsWih2, srcWih2, tid);
    stage_slice(ldsWhh2, srcWhh2, tid);
#endif
    __syncthreads();

    // hint the next spin's slice toward L2 while we grind this one
    if (spin + 1 < NSPIN && tid < 16) {
      const char* nxt = (const char*)(whh0b + ((size_t)(spin + 1) * HID + wg * ROWS_PER_WG) * HID);
      __builtin_prefetch(nxt + (size_t)tid * 4096, 0, 1);   // -> global_prefetch_b8
    }

    // per-spin, per-lane constants (row is fixed for a lane)
    const float b0   = bih0[spin * HID + row] + bhh0[spin * HID + row];
    const float b1   = bihL[(spin * 2 + 0) * HID + row] + bhhL[(spin * 2 + 0) * HID + row];
    const float b2   = bihL[(spin * 2 + 1) * HID + row] + bhhL[(spin * 2 + 1) * HID + row];
    const float w0c0 = Wih0[((size_t)spin * HID + row) * 2 + 0];
    const float w0c1 = Wih0[((size_t)spin * HID + row) * 2 + 1];

    for (int t = 0; t < SEQ; ++t) {
      const int pp = t & 1;       // write buffer
      const int pr = pp ^ 1;      // read (previous step) buffer

      // ---- layer 0: h0 = tanh(Wih0[:,x] + Whh0 @ h0_prev + b) -------------
      {
        v8f c = {};
        c = matvec_tile_acc(hbuf + (0 * 2 + pr) * HID, ldsWhh0, wave * 16, lane, c);
        float val = c[0] + b0;
        if (spin > 0) {
          int cp = samples[t * NSPIN + spin - 1];     // one-hot column gather
          val += cp ? w0c1 : w0c0;
        }
        if (lane < 16) hbuf[(0 * 2 + pp) * HID + row] = f2bf(tanhf(val));
      }
      cluster_sync();

      // ---- layer 1: h1 = tanh(Wih1 @ h0 + Whh1 @ h1_prev + b) -------------
      {
        v8f c = {};
        c = matvec_tile_acc(hbuf + (0 * 2 + pp) * HID, ldsWih1, wave * 16, lane, c);
        c = matvec_tile_acc(hbuf + (1 * 2 + pr) * HID, ldsWhh1, wave * 16, lane, c);
        if (lane < 16) hbuf[(1 * 2 + pp) * HID + row] = f2bf(tanhf(c[0] + b1));
      }
      cluster_sync();

      // ---- layer 2 --------------------------------------------------------
      {
        v8f c = {};
        c = matvec_tile_acc(hbuf + (1 * 2 + pp) * HID, ldsWih2, wave * 16, lane, c);
        c = matvec_tile_acc(hbuf + (2 * 2 + pr) * HID, ldsWhh2, wave * 16, lane, c);
        if (lane < 16) hbuf[(2 * 2 + pp) * HID + row] = f2bf(tanhf(c[0] + b2));
      }
      cluster_sync();

      // ---- dense head + softmax on WG0/wave0 ------------------------------
      if (wg == 0 && wave == 0) {
        const u16* h2 = hbuf + (2 * 2 + pp) * HID;
        float s0 = 0.f, s1 = 0.f;
        for (int k = lane; k < HID; k += 32) {
          float hv = bf2f(h2[k]);
          s0 += hv * Wd[((size_t)spin * 2 + 0) * HID + k];
          s1 += hv * Wd[((size_t)spin * 2 + 1) * HID + k];
        }
        for (int o = 16; o; o >>= 1) {
          s0 += __shfl_xor(s0, o, 32);
          s1 += __shfl_xor(s1, o, 32);
        }
        if (lane == 0) {
          float l0 = s0 + bd[spin * 2 + 0], l1 = s1 + bd[spin * 2 + 1];
          float m  = fmaxf(l0, l1);
          float e0 = __expf(l0 - m), e1 = __expf(l1 - m);
          int cls  = samples[t * NSPIN + spin];
          pbuf[t * NSPIN + spin] = (cls ? e1 : e0) / (e0 + e1);
        }
      }
    }
  }
}

// ---------------------------------------------------------------------------
__global__ void prod_over_spins(const float* __restrict__ pbuf,
                                float* __restrict__ out) {
  int t = blockIdx.x * blockDim.x + threadIdx.x;
  if (t < SEQ) {
    float p = 1.f;
#pragma unroll
    for (int i = 0; i < NSPIN; ++i) p *= pbuf[t * NSPIN + i];
    out[t] = p;
  }
}

// ---------------------------------------------------------------------------
extern "C" void kernel_launch(void* const* d_in, const int* in_sizes, int n_in,
                              void* d_out, int out_size, void* d_ws, size_t ws_size,
                              hipStream_t stream) {
  const int*   samples = (const int*)  d_in[0];
  const float* Wih0    = (const float*)d_in[1];
  const float* Whh0    = (const float*)d_in[2];
  const float* bih0    = (const float*)d_in[3];
  const float* bhh0    = (const float*)d_in[4];
  const float* Wih     = (const float*)d_in[5];
  const float* Whh     = (const float*)d_in[6];
  const float* bihL    = (const float*)d_in[7];
  const float* bhhL    = (const float*)d_in[8];
  const float* Wd      = (const float*)d_in[9];
  const float* bd      = (const float*)d_in[10];
  float* out = (float*)d_out;

  char* ws = (char*)d_ws;
  const size_t OFF_WIH  = 16777216ull;               // 16 MB
  const size_t OFF_WHH  = OFF_WIH + 33554432ull;     // 48 MB
  const size_t OFF_HBUF = 83886080ull;               // 80 MB
  const size_t OFF_PBUF = OFF_HBUF + 8192ull;
  u16*   whh0b = (u16*)(ws);
  u16*   wihb  = (u16*)(ws + OFF_WIH);
  u16*   whhb  = (u16*)(ws + OFF_WHH);
  u16*   hbuf  = (u16*)(ws + OFF_HBUF);
  float* pbuf  = (float*)(ws + OFF_PBUF);

  // 1) f32 -> bf16 weight conversion (41,943,040 elements)
  convert_weights_bf16<<<163840, 256, 0, stream>>>(Whh0, Wih, Whh, whh0b, wihb, whhb);
  // 2) zero the carried hidden state (state0 = 0 every launch)
  zero_hbuf<<<12, 256, 0, stream>>>(hbuf);
  // 3) the sequential RNN chain: 16-WG cluster, 160 KB LDS per WG
  rnn_persistent<<<dim3(NWG), dim3(64), 5 * SLICE_HALVES * sizeof(u16), stream>>>(
      samples, Wih0, bih0, bhh0, bihL, bhhL, Wd, bd,
      whh0b, wihb, whhb, hbuf, pbuf);
  // 4) out[t] = prod_i pbuf[t][i]
  prod_over_spins<<<SEQ / 256, 256, 0, stream>>>(pbuf, out);
}
